// LSTM2LINEAR_70188355551444
// MI455X (gfx1250) — matrix operand order repaired and checked by reading, not compiled, not run on hardware
//
#include <hip/hip_runtime.h>
#include <hip/hip_bf16.h>

typedef __attribute__((ext_vector_type(16))) __bf16 v16bf;
typedef __attribute__((ext_vector_type(2)))  __bf16 v2bf;
typedef __attribute__((ext_vector_type(8)))  float  v8f;
typedef __attribute__((ext_vector_type(8)))  int    v8i;

namespace {
constexpr int kB = 8192;
constexpr int kT = 1024;
constexpr int kH = 10;      // hidden
constexpr int kG = 4 * kH;  // 40 gate rows
}

// lane <-> lane^16 swap. v_permlanex16_b32 (VALU, no DS counter / waits).
__device__ __forceinline__ int xswap16_i(int v) {
#if __has_builtin(__builtin_amdgcn_permlanex16)
    return __builtin_amdgcn_permlanex16(v, v, 0x76543210, 0xfedcba98, false, false);
#else
    return __shfl_xor(v, 16, 32);
#endif
}
__device__ __forceinline__ float xswap16_f(float v) {
    return __int_as_float(xswap16_i(__float_as_int(v)));
}

__device__ __forceinline__ float tanh_fast(float x) {
#if __has_builtin(__builtin_amdgcn_tanhf)
    return __builtin_amdgcn_tanhf(x);   // gfx1250 v_tanh_f32
#else
    float e = __expf(-2.0f * x);
    return __builtin_fmaf(-2.0f, __builtin_amdgcn_rcpf(1.0f + e), 1.0f);
#endif
}
__device__ __forceinline__ float sigm(float x) {
#if __has_builtin(__builtin_amdgcn_tanhf)
    // sigmoid(x) = 0.5*tanh(x/2) + 0.5 : one trans op instead of exp+rcp
    return __builtin_fmaf(0.5f, __builtin_amdgcn_tanhf(0.5f * x), 0.5f);
#else
    return __builtin_amdgcn_rcpf(1.0f + __expf(-x));
#endif
}
__device__ __forceinline__ unsigned bf16_bits(float f) {  // RNE f32->bf16
    unsigned u = __float_as_uint(f);
    return (u + 0x7fffu + ((u >> 16) & 1u)) >> 16;
}
__device__ __forceinline__ int pack_bf16_pair(float lo, float hi_v) {
#if __has_builtin(__builtin_amdgcn_cvt_pk_bf16_f32)
    v2bf pk = __builtin_amdgcn_cvt_pk_bf16_f32(lo, hi_v);  // v_cvt_pk_bf16_f32
    return __builtin_bit_cast(int, pk);
#else
    return (int)(bf16_bits(lo) | (bf16_bits(hi_v) << 16));
#endif
}

// One wave32 handles a 16-row batch tile. gates.T[48,16] = A(W_hh)[48x32] x B(h.T)[32x16] + C
__global__ __launch_bounds__(128)
void lstm2linear_wmma(const float* __restrict__ x,
                      const float* __restrict__ W_ih,
                      const float* __restrict__ W_hh,
                      const float* __restrict__ b_ih,
                      const float* __restrict__ b_hh,
                      const float* __restrict__ W_lin,
                      const float* __restrict__ b_lin,
                      float* __restrict__ out)
{
    const int lane = threadIdx.x & 31;
    const int wid  = blockIdx.x * 4 + (threadIdx.x >> 5);   // 0..511
    const int col  = lane & 15;                             // batch column in tile
    const bool hi  = lane >= 16;
    const int bat  = wid * 16 + col;

    // ---- constant A fragments: W_hh[40,10] -> 3 bf16 16x32 A-layout tiles (K padded) ----
    // lanes 0-15: row M=16t+col, v[r]=K(2r,2r+1) r<5 ; lanes 16-31: 0; rest 0.
    v16bf A[3];
    #pragma unroll
    for (int t = 0; t < 3; ++t) {
        v8i ai = {0, 0, 0, 0, 0, 0, 0, 0};
        int m = 16 * t + col;
        if (m < kG) {
            const float* wr = W_hh + m * kH;
            if (!hi) {
                #pragma unroll
                for (int r = 0; r < 5; ++r)
                    ai[r] = (int)(bf16_bits(wr[2 * r]) | (bf16_bits(wr[2 * r + 1]) << 16));
            }
        }
        A[t] = __builtin_bit_cast(v16bf, ai);
    }

    // ---- per-lane C-layout constants: gate G = 16t + 8*hi + r ----
    float biasf[3][8], wihf[3][8];
    #pragma unroll
    for (int t = 0; t < 3; ++t) {
        int gb = 16 * t + (hi ? 8 : 0);
        #pragma unroll
        for (int r = 0; r < 8; ++r) {
            int G = gb + r;
            biasf[t][r] = (G < kG) ? (b_ih[G] + b_hh[G]) : 0.0f;
            wihf[t][r]  = (G < kG) ? W_ih[G] : 0.0f;
        }
    }

    // state: each lane owns hidden units j = 2k + hi, k=0..4
    float c[5] = {0.f, 0.f, 0.f, 0.f, 0.f};
    v8i bfrag = {0, 0, 0, 0, 0, 0, 0, 0};   // h starts at zero

    const float* xp = x + (size_t)bat * kT;
    float4 xc = *reinterpret_cast<const float4*>(xp);

    for (int t = 0; t < kT; t += 4) {
        // unconditional wrapped prefetch (last iter loads t=0 again; value unused)
        const int tn = (t + 4) & (kT - 1);
        float4 xn = *reinterpret_cast<const float4*>(xp + tn);
        float xs[4] = {xc.x, xc.y, xc.z, xc.w};

        #pragma unroll
        for (int s = 0; s < 4; ++s) {
            const float xv = xs[s];
            v16bf Bf = __builtin_bit_cast(v16bf, bfrag);

            v8f c0, c1, c2;
            #pragma unroll
            for (int r = 0; r < 8; ++r) {
                c0[r] = __builtin_fmaf(xv, wihf[0][r], biasf[0][r]);
                c1[r] = __builtin_fmaf(xv, wihf[1][r], biasf[1][r]);
                c2[r] = __builtin_fmaf(xv, wihf[2][r], biasf[2][r]);
            }

            v8f d0 = __builtin_amdgcn_wmma_f32_16x16x32_bf16(false, A[0], false, Bf, (short)0, c0, false, false);
            v8f d1 = __builtin_amdgcn_wmma_f32_16x16x32_bf16(false, A[1], false, Bf, (short)0, c1, false, false);
            v8f d2 = __builtin_amdgcn_wmma_f32_16x16x32_bf16(false, A[2], false, Bf, (short)0, c2, false, false);

            // cross-half copies (lane ^ 16) of the D registers we need (pure VALU permlane)
            float s0[8], s1[8], s2[8];
            #pragma unroll
            for (int r = 0; r < 8; ++r) {
                s0[r] = xswap16_f(d0[r]);
                s1[r] = xswap16_f(d1[r]);
            }
            #pragma unroll
            for (int r = 1; r < 8; r += 2) s2[r] = xswap16_f(d2[r]);

            // elementwise LSTM update; lane's unit j = 2k + hi
            float hf[5];
            #pragma unroll
            for (int k = 0; k < 5; ++k) {
                float ik, fk, gk, ok;
                if (k < 4) ik = hi ? s0[2 * k + 1] : d0[2 * k];
                else       ik = hi ? d0[1]         : s0[0];
                if (k < 3) fk = hi ? d0[2 * k + 3] : s0[2 * k + 2];
                else       fk = hi ? s1[2 * k - 5] : d1[2 * k - 6];
                if (k < 2) gk = hi ? s1[2 * k + 5] : d1[2 * k + 4];
                else       gk = hi ? d1[2 * k - 3] : s1[2 * k - 4];
                if (k == 0) ok = hi ? d1[7]        : s1[6];
                else        ok = hi ? s2[2 * k - 1] : d2[2 * k - 2];

                float cn = __builtin_fmaf(sigm(fk), c[k], sigm(ik) * tanh_fast(gk));
                c[k] = cn;
                hf[k] = sigm(ok) * tanh_fast(cn);
            }

            // rebuild B fragment: lanes 0-15 pack (h_even, h_odd) -> bf16x2; lanes 16-31 (K>=16) zero
            #pragma unroll
            for (int k = 0; k < 5; ++k) {
                float p = xswap16_f(hf[k]);
                bfrag[k] = hi ? 0 : pack_bf16_pair(hf[k], p);
            }
            bfrag[5] = 0; bfrag[6] = 0; bfrag[7] = 0;
        }
        xc = xn;
    }

    // final projection: out[b] = dot(c_T[b], W_lin) + b_lin
    float acc = 0.f;
    #pragma unroll
    for (int k = 0; k < 5; ++k)
        acc = __builtin_fmaf(c[k], W_lin[2 * k + (hi ? 1 : 0)], acc);
    acc += xswap16_f(acc);
    if (!hi) out[bat] = acc + b_lin[0];
}

extern "C" void kernel_launch(void* const* d_in, const int* in_sizes, int n_in,
                              void* d_out, int out_size, void* d_ws, size_t ws_size,
                              hipStream_t stream) {
    (void)in_sizes; (void)n_in; (void)d_ws; (void)ws_size; (void)out_size;
    const float* x     = (const float*)d_in[0];
    const float* W_ih  = (const float*)d_in[1];
    const float* W_hh  = (const float*)d_in[2];
    const float* b_ih  = (const float*)d_in[3];
    const float* b_hh  = (const float*)d_in[4];
    const float* W_lin = (const float*)d_in[5];
    const float* b_lin = (const float*)d_in[6];
    float* out = (float*)d_out;

    // 512 waves (B/16), 4 waves32 per block -> 128 blocks x 128 threads
    dim3 grid(kB / 16 / 4), block(128);
    lstm2linear_wmma<<<grid, block, 0, stream>>>(x, W_ih, W_hh, b_ih, b_hh, W_lin, b_lin, out);
}
